// EntropyMaskGate_60327110640100
// MI455X (gfx1250) — matrix operand order repaired and checked by reading, not compiled, not run on hardware
//
#include <hip/hip_runtime.h>
#include <hip/hip_bf16.h>

// ---------------------------------------------------------------------------
// EntropyMaskGate on MI455X (gfx1250, wave32, WMMA).
//  conv1 (1x1, 256->64) : bf16 WMMA GEMM, f32 accum, exact GELU epilogue
//  conv2 (3x3 grouped)  : direct VALU (K=72, too small for WMMA)
//  conv3 (1x1, 64->256) : bf16 WMMA GEMM -> entropy_scores (f32, to d_out)
//  mask                 : 2x2 pool + per-row top-256-of-1024 hard mask
// Forward value of (stop_grad(hard) - stop_grad(soft) + soft) == hard.
// ---------------------------------------------------------------------------

typedef __attribute__((ext_vector_type(16))) __bf16 v16bf;
typedef __attribute__((ext_vector_type(8)))  __bf16 v8bf;
typedef __attribute__((ext_vector_type(8)))  float  v8f;

#define BB    16
#define CCH   256
#define HH    64
#define WWD   64
#define HWE   4096          // 64*64
#define MIDC  64
#define NBLK  1024          // 32*32 pooled blocks per (b,c) row
#define KEEPN 256           // round(1024 * 0.25)

#define XSTR  264           // padded LDS stride (bf16 elems) for X tiles
#define WSTR  264           // padded LDS stride for W1
#define W3STR 72            // padded LDS stride for conv3 (K=64)

// ---- bf16 <-> f32 via bit ops (keeps __bf16 device-only) -------------------
__device__ __forceinline__ unsigned short f2bfu(float f) {
  union { float f; unsigned u; } in; in.f = f;
  unsigned r = (in.u + 0x7FFFu + ((in.u >> 16) & 1u)) >> 16;   // RNE
  return (unsigned short)r;
}
__device__ __forceinline__ float bfu2f(unsigned short s) {
  union { unsigned u; float f; } out; out.u = ((unsigned)s) << 16;
  return out.f;
}
__device__ __forceinline__ __bf16 f2bf(float f) {
  return __builtin_bit_cast(__bf16, f2bfu(f));
}
__device__ __forceinline__ float gelu_exact(float x) {
  return 0.5f * x * (1.0f + erff(x * 0.70710678118654752f));
}

// ===========================================================================
// conv1: y[b,m,p] = gelu( b1[m] + sum_c w1[m,c] * feat[b,c,p] ), m<64, c<256
// Workgroup: 8 waves; tile = (batch b, 128 pixels, all 64 M).
// Wave (wm,wn): M rows 16*wm..+15, N cols 64*wn..+63 (4 WMMA tiles).
// ===========================================================================
__global__ __launch_bounds__(256) void conv1_gelu_wmma(
    const float* __restrict__ feat, const float* __restrict__ w1,
    const float* __restrict__ b1, unsigned short* __restrict__ h1,
    const int* __restrict__ enabled)
{
  if (*enabled == 0) return;

  __shared__ __bf16 sW[MIDC * WSTR];   // [m][k]
  __shared__ __bf16 sX[128 * XSTR];    // [p][k]  (transposed tile)

  const int tid = threadIdx.x;
  const int b   = blockIdx.x >> 5;         // / 32 tiles per batch
  const int p0  = (blockIdx.x & 31) << 7;  // * 128 pixels

  // stage W1 (64x256 f32 -> bf16), coalesced
  for (int i = 0; i < 64; ++i)
    sW[i * WSTR + tid] = f2bf(w1[i * CCH + tid]);

  // stage X tile transposed: sX[p][c]; reads coalesced along pixels
  {
    const int pl = tid & 127;
    const int c0 = (tid >> 7) * 128;
    const float* src = feat + (size_t)b * CCH * HWE + p0 + pl;
    for (int k = 0; k < 128; ++k) {
      int c = c0 + k;
      if (k + 8 < 128) __builtin_prefetch(src + (size_t)(c + 8) * HWE, 0, 0);
      sX[pl * XSTR + c] = f2bf(src[(size_t)c * HWE]);
    }
  }
  __syncthreads();

  const int lane = tid & 31;
  const int wave = tid >> 5;
  const int half = lane >> 4;     // 0: lanes 0-15, 1: lanes 16-31
  const int l16  = lane & 15;
  const int m0   = (wave & 3) << 4;
  const int n0   = (wave >> 2) << 6;

  v8f zero = {0.f, 0.f, 0.f, 0.f, 0.f, 0.f, 0.f, 0.f};
  v8f acc[4] = {zero, zero, zero, zero};

  for (int kk = 0; kk < CCH; kk += 32) {
    // A fragment (16x32 bf16): lane<16 holds K kk..kk+7 & kk+16..kk+23 of row
    v16bf a;
    {
      const __bf16* pa = &sW[(m0 + l16) * WSTR + kk + half * 8];
      v8bf lo = *(const v8bf*)(pa);
      v8bf hi = *(const v8bf*)(pa + 16);
      for (int e = 0; e < 8; ++e) { a[e] = lo[e]; a[e + 8] = hi[e]; }
    }
    for (int j = 0; j < 4; ++j) {
      // B fragment (32x16): lane<16 holds K kk..kk+7 & kk+16..kk+23 of column
      v16bf bm;
      {
        const __bf16* pb = &sX[(n0 + j * 16 + l16) * XSTR + kk + half * 8];
        v8bf lo = *(const v8bf*)(pb);
        v8bf hi = *(const v8bf*)(pb + 16);
        for (int e = 0; e < 8; ++e) { bm[e] = lo[e]; bm[e + 8] = hi[e]; }
      }
      acc[j] = __builtin_amdgcn_wmma_f32_16x16x32_bf16(
          false, a, false, bm, (short)0, acc[j], false, false);
    }
  }

  // epilogue: C/D layout -> VGPR r holds M=m0+8*half+r, N=lane%16
  for (int j = 0; j < 4; ++j) {
    const int p = p0 + n0 + j * 16 + l16;
    for (int r = 0; r < 8; ++r) {
      const int m = m0 + half * 8 + r;
      float v = acc[j][r] + b1[m];
      h1[((size_t)b * MIDC + m) * HWE + p] = f2bfu(gelu_exact(v));
    }
  }
}

// ===========================================================================
// conv2: grouped 3x3, 8 groups of 8 channels; direct VALU (72 MACs/output)
// ===========================================================================
__global__ __launch_bounds__(256) void conv2_grouped_gelu(
    const unsigned short* __restrict__ h1, const float* __restrict__ w2,
    const float* __restrict__ b2, unsigned short* __restrict__ h2,
    const int* __restrict__ enabled)
{
  if (*enabled == 0) return;
  const int idx = blockIdx.x * 256 + threadIdx.x;   // B*MID*HW threads
  const int p = idx & (HWE - 1);
  const int m = (idx >> 12) & (MIDC - 1);
  const int b = idx >> 18;
  const int x = p & 63, y = p >> 6;
  const int g = m >> 3;

  float sum = b2[m];
  const unsigned short* base = h1 + ((size_t)b * MIDC + g * 8) * HWE;
  const float* wbase = w2 + m * 8 * 9;
  for (int ci = 0; ci < 8; ++ci) {
    const unsigned short* hp = base + (size_t)ci * HWE;
    const float* wp = wbase + ci * 9;
    for (int dy = 0; dy < 3; ++dy) {
      int yy = y + dy - 1;
      if ((unsigned)yy >= 64u) continue;
      for (int dx = 0; dx < 3; ++dx) {
        int xx = x + dx - 1;
        if ((unsigned)xx >= 64u) continue;
        sum += bfu2f(hp[yy * 64 + xx]) * wp[dy * 3 + dx];
      }
    }
  }
  h2[idx] = f2bfu(gelu_exact(sum));
}

// ===========================================================================
// conv3: ent[b,mc,p] = b3[mc] + sum_k w3[mc,k]*h2[b,k,p], mc<256, k<64
// Workgroup: 8 waves; tile = (batch b, 64 pixels, all 256 M).
// Wave w: M rows 32*w..+31 (2 M-blocks) x 4 N-blocks. K-loop: 2 x 32.
// ===========================================================================
__global__ __launch_bounds__(256) void conv3_wmma(
    const unsigned short* __restrict__ h2, const float* __restrict__ w3,
    const float* __restrict__ b3, float* __restrict__ ent,
    const int* __restrict__ enabled)
{
  if (*enabled == 0) return;

  __shared__ __bf16 sW[CCH * W3STR];   // [m=256][k=64]
  __shared__ __bf16 sH[64 * W3STR];    // [p=64][k=64]

  const int tid = threadIdx.x;
  const int b   = blockIdx.x >> 6;         // / 64 tiles per batch
  const int p0  = (blockIdx.x & 63) << 6;  // * 64 pixels

  for (int i = 0; i < 64; ++i) {
    int idx = i * 256 + tid;
    sW[(idx >> 6) * W3STR + (idx & 63)] = f2bf(w3[idx]);
  }
  {
    const int pl = tid & 63;
    const int k0 = (tid >> 6) << 4;
    const unsigned short* src = h2 + (size_t)b * MIDC * HWE + p0 + pl;
    for (int i = 0; i < 16; ++i) {
      int k = k0 + i;
      sH[pl * W3STR + k] = __builtin_bit_cast(__bf16, src[(size_t)k * HWE]);
    }
  }
  __syncthreads();

  const int lane = tid & 31;
  const int wave = tid >> 5;
  const int half = lane >> 4;
  const int l16  = lane & 15;
  const int m0   = wave << 5;   // 32 M rows per wave

  v8f zero = {0.f, 0.f, 0.f, 0.f, 0.f, 0.f, 0.f, 0.f};
  v8f acc[2][4] = {{zero, zero, zero, zero}, {zero, zero, zero, zero}};

  for (int kk = 0; kk < 64; kk += 32) {
    v16bf a[2];
    for (int s = 0; s < 2; ++s) {
      const __bf16* pa = &sW[(m0 + s * 16 + l16) * W3STR + kk + half * 8];
      v8bf lo = *(const v8bf*)(pa);
      v8bf hi = *(const v8bf*)(pa + 16);
      for (int e = 0; e < 8; ++e) { a[s][e] = lo[e]; a[s][e + 8] = hi[e]; }
    }
    for (int j = 0; j < 4; ++j) {
      v16bf bm;
      {
        const __bf16* pb = &sH[(j * 16 + l16) * W3STR + kk + half * 8];
        v8bf lo = *(const v8bf*)(pb);
        v8bf hi = *(const v8bf*)(pb + 16);
        for (int e = 0; e < 8; ++e) { bm[e] = lo[e]; bm[e + 8] = hi[e]; }
      }
      acc[0][j] = __builtin_amdgcn_wmma_f32_16x16x32_bf16(
          false, a[0], false, bm, (short)0, acc[0][j], false, false);
      acc[1][j] = __builtin_amdgcn_wmma_f32_16x16x32_bf16(
          false, a[1], false, bm, (short)0, acc[1][j], false, false);
    }
  }

  for (int s = 0; s < 2; ++s)
    for (int j = 0; j < 4; ++j) {
      const int p = p0 + j * 16 + l16;
      for (int r = 0; r < 8; ++r) {
        const int m = m0 + s * 16 + half * 8 + r;
        ent[((size_t)b * CCH + m) * HWE + p] = acc[s][j][r] + b3[m];
      }
    }
}

// ===========================================================================
// mask: per (b,c) row -> 2x2 avg pool to 1024 blocks, keep 256 highest
// importance (= lowest pooled). Forward mask == hard top-k mask, 2x2-upsampled.
// Tie-break (equal value -> lower index wins) matches lax.top_k.
// Disabled path: mask=1, entropy=0.
// ===========================================================================
__global__ __launch_bounds__(256) void topk_mask(
    float* __restrict__ mask, float* __restrict__ ent,
    const int* __restrict__ enabled)
{
  const int tid = threadIdx.x;
  const int row = blockIdx.x;                 // b*C + c (4096 rows)
  float* mrow = mask + (size_t)row * HWE;
  float* erow = ent  + (size_t)row * HWE;

  if (*enabled == 0) {
    for (int i = 0; i < 16; ++i) {
      mrow[i * 256 + tid] = 1.0f;
      erow[i * 256 + tid] = 0.0f;
    }
    return;
  }

  __shared__ float simp[NBLK];
  for (int q = 0; q < 4; ++q) {
    const int i  = tid * 4 + q;
    const int by = i >> 5, bx = i & 31;
    const float* pp = erow + (by * 2) * WWD + bx * 2;
    simp[i] = -0.25f * (pp[0] + pp[1] + pp[WWD] + pp[WWD + 1]);  // importance
  }
  __syncthreads();

  for (int q = 0; q < 4; ++q) {
    const int i = tid * 4 + q;
    const float v = simp[i];
    int rank = 0;
    for (int j = 0; j < NBLK; ++j) {
      const float u = simp[j];
      rank += (u > v) || (u == v && j < i);
    }
    const float mv = (rank < KEEPN) ? 1.0f : 0.0f;
    const int by = i >> 5, bx = i & 31;
    float* mp = mrow + (by * 2) * WWD + bx * 2;
    mp[0] = mv; mp[1] = mv; mp[WWD] = mv; mp[WWD + 1] = mv;
  }
}

// ===========================================================================
extern "C" void kernel_launch(void* const* d_in, const int* in_sizes, int n_in,
                              void* d_out, int out_size, void* d_ws, size_t ws_size,
                              hipStream_t stream) {
  const float* feat    = (const float*)d_in[0];
  const float* w1      = (const float*)d_in[1];
  const float* b1      = (const float*)d_in[2];
  const float* w2      = (const float*)d_in[3];
  const float* b2      = (const float*)d_in[4];
  const float* w3      = (const float*)d_in[5];
  const float* b3      = (const float*)d_in[6];
  const int*   enabled = (const int*)d_in[7];

  float* mask = (float*)d_out;
  float* ent  = (float*)d_out + (size_t)BB * CCH * HWE;

  unsigned short* h1 = (unsigned short*)d_ws;                 // B*MID*HW bf16
  unsigned short* h2 = h1 + (size_t)BB * MIDC * HWE;          // B*MID*HW bf16

  conv1_gelu_wmma   <<<BB * 32,                 256, 0, stream>>>(feat, w1, b1, h1, enabled);
  conv2_grouped_gelu<<<(BB * MIDC * HWE) / 256, 256, 0, stream>>>(h1, w2, b2, h2, enabled);
  conv3_wmma        <<<BB * 64,                 256, 0, stream>>>(h2, w3, b3, ent, enabled);
  topk_mask         <<<BB * CCH,                256, 0, stream>>>(mask, ent, enabled);
}
